// MultiHeadAttention_11854109737154
// MI455X (gfx1250) — matrix operand order repaired and checked by reading, not compiled, not run on hardware
//
#include <hip/hip_runtime.h>
#include <hip/hip_bf16.h>

#define DEV __device__ __forceinline__

typedef __attribute__((ext_vector_type(16))) __bf16 v16bf;
typedef __attribute__((ext_vector_type(8)))  float  v8f;

constexpr int Bb = 2, Hh = 16, Ll = 2048, Dd = 2048, HD = 128;
constexpr int GSTR = 72;   // GEMM LDS row stride (ushorts): 64 data + 8 pad, 16B aligned
constexpr int KSTR = 136;  // flash K-tile stride: 128 + 8 pad
constexpr int VSTR = 40;   // flash V-tile / P-buffer stride: 32 + 8 pad
constexpr float SCALE = 0.08838834764831845f;  // 1/sqrt(128)

// ---------- helpers ----------
DEV unsigned short f2bf(float f) {             // f32 -> bf16, round-to-nearest-even
  unsigned int u = __float_as_uint(f);
  u += 0x7FFFu + ((u >> 16) & 1u);
  return (unsigned short)(u >> 16);
}
DEV unsigned int pack2(float a, float b) {
  return (unsigned int)f2bf(a) | ((unsigned int)f2bf(b) << 16);
}

union FragCvt { uint4 u[2]; v16bf v; };

// A-fragment: elements 0..7 from p0 (K kb..kb+7), 8..15 from p1 (K kb+16..kb+23)
DEV v16bf frag_a(const unsigned short* p0, const unsigned short* p1) {
  FragCvt c; c.u[0] = *(const uint4*)p0; c.u[1] = *(const uint4*)p1; return c.v;
}
// B-fragment: elements 0..7 from p (K kb..kb+7), 8..15 from p+16 (K kb+16..kb+23)
DEV v16bf frag_b(const unsigned short* p) {
  FragCvt c; c.u[0] = *(const uint4*)p; c.u[1] = *(const uint4*)(p + 16); return c.v;
}
DEV v8f wmma_bf16(v16bf a, v16bf b, v8f c) {
  return __builtin_amdgcn_wmma_f32_16x16x32_bf16(false, a, false, b, (short)0, c, false, false);
}

// ---------- 16-lane-row reductions: v_permlane16_b32 (VALU-only, no LDS) ----------
#if __has_builtin(__builtin_amdgcn_permlane16)
#define HAVE_PERMLANE16 1
DEV float permx(float v, unsigned s0, unsigned s1) {
  unsigned u = __float_as_uint(v);
  return __uint_as_float(__builtin_amdgcn_permlane16(u, u, s0, s1, false, false));
}
DEV float row16_max(float v) {
  v = fmaxf(v, permx(v, 0x67452301u, 0xEFCDAB89u));   // xor 1
  v = fmaxf(v, permx(v, 0x54761032u, 0xDCFE98BAu));   // xor 2
  v = fmaxf(v, permx(v, 0x32107654u, 0xBA98FEDCu));   // xor 4
  v = fmaxf(v, permx(v, 0xFEDCBA98u, 0x76543210u));   // xor 8
  return v;
}
DEV float row16_sum(float v) {
  v += permx(v, 0x67452301u, 0xEFCDAB89u);
  v += permx(v, 0x54761032u, 0xDCFE98BAu);
  v += permx(v, 0x32107654u, 0xBA98FEDCu);
  v += permx(v, 0xFEDCBA98u, 0x76543210u);
  return v;
}
#else
DEV float row16_max(float v) {
#pragma unroll
  for (int m = 1; m < 16; m <<= 1) v = fmaxf(v, __shfl_xor(v, m, 32));
  return v;
}
DEV float row16_sum(float v) {
#pragma unroll
  for (int m = 1; m < 16; m <<= 1) v += __shfl_xor(v, m, 32);
  return v;
}
#endif

// ---------- CDNA5 async global->LDS copy (ASYNCcnt-tracked) ----------
DEV unsigned lds_off(const void* p) {          // low 32 bits of generic LDS addr = LDS offset
  return (unsigned)(unsigned long long)p;
}
DEV void async_g2l_b128(unsigned loff, const unsigned short* g) {
  asm volatile("global_load_async_to_lds_b128 %0, %1, off"
               :: "v"(loff), "v"(g) : "memory");
}
DEV void wait_async0() { asm volatile("s_wait_asynccnt 0x0" ::: "memory"); }

// f32x16 -> bf16x16 convert + LDS store (for staging f32 operands)
DEV void stage_f32(const float* g, unsigned short* l) {
  const float4* g4 = (const float4*)g;
  float4 f0 = g4[0], f1 = g4[1], f2 = g4[2], f3 = g4[3];
  uint4 s0 = { pack2(f0.x, f0.y), pack2(f0.z, f0.w), pack2(f1.x, f1.y), pack2(f1.z, f1.w) };
  uint4 s1 = { pack2(f2.x, f2.y), pack2(f2.z, f2.w), pack2(f3.x, f3.y), pack2(f3.z, f3.w) };
  *(uint4*)l = s0; *(uint4*)(l + 8) = s1;
}

// ============================================================================
// GEMM: C[M=4096, N=2048] = A[4096,2048] * W[2048,2048]^T + bias
// 128x128 block tile, 8 waves x (64x32), K-step 64 (16 WMMAs between barriers)
// EPI 0: bf16 out [B,H,L,HD] with rotary (cos-sin) on first half of head dim (Q,K)
// EPI 1: bf16 out transposed [B,H,HD,L]                                      (V)
// EPI 2: f32 out [B*L, D] + bias                                             (O)
// ============================================================================
template<bool A_BF16, int EPI>
__global__ __launch_bounds__(256)
void gemm_wmma(const void* __restrict__ Aptr, const float* __restrict__ W,
               const float* __restrict__ bias, void* __restrict__ Out,
               const float* __restrict__ cosp, const float* __restrict__ sinp)
{
  __shared__ __align__(16) unsigned short lA[128 * GSTR];
  __shared__ __align__(16) unsigned short lB[128 * GSTR];

  const int tid  = threadIdx.x;
  const int lane = tid & 31;
  const int wave = tid >> 5;
  const int wm = wave >> 2;        // 0..1  (64-row slab)
  const int wn = wave & 3;         // 0..3  (32-col slab)
  const int m0 = blockIdx.y * 128;
  const int n0 = blockIdx.x * 128;

  const int l15 = lane & 15;
  const int hl8 = (lane & 16) ? 8 : 0;
  const int akb = (lane & 16) ? 8 : 0;   // A-fragment K base
  const int bkb = (lane & 16) ? 8 : 0;   // B-fragment K base

  v8f zero = {};
  v8f acc[4][2];
#pragma unroll
  for (int i = 0; i < 4; i++)
#pragma unroll
    for (int j = 0; j < 2; j++) acc[i][j] = zero;

  for (int k0 = 0; k0 < Dd; k0 += 64) {
    __syncthreads();
    // ---- stage A tile (128 x 64 bf16) ----
    if (A_BF16) {
      // direct async copy: 1024 16B-chunks, 4 per thread
#pragma unroll
      for (int i = 0; i < 4; i++) {
        int c = tid + 256 * i;
        int row = c >> 3, c8 = c & 7;
        const unsigned short* g = (const unsigned short*)Aptr +
                                  (size_t)(m0 + row) * Dd + k0 + c8 * 8;
        async_g2l_b128(lds_off(&lA[row * GSTR + c8 * 8]), g);
      }
    } else {
      // f32 -> bf16 convert: 512 16-elem chunks, 2 per thread
#pragma unroll
      for (int i = 0; i < 2; i++) {
        int c = tid + 256 * i;
        int row = c >> 2, c4 = c & 3;
        const float* g = (const float*)Aptr + (size_t)(m0 + row) * Dd + k0 + c4 * 16;
        stage_f32(g, &lA[row * GSTR + c4 * 16]);
        if (k0 + 64 < Dd) __builtin_prefetch(g + 64);
      }
    }
    // ---- stage B tile (weights, always f32) ----
#pragma unroll
    for (int i = 0; i < 2; i++) {
      int c = tid + 256 * i;
      int row = c >> 2, c4 = c & 3;
      const float* g = W + (size_t)(n0 + row) * Dd + k0 + c4 * 16;
      stage_f32(g, &lB[row * GSTR + c4 * 16]);
      if (k0 + 64 < Dd) __builtin_prefetch(g + 64);
    }
    if (A_BF16) wait_async0();
    __syncthreads();

    // ---- 16 WMMAs per wave (two K-chunks of 32) ----
#pragma unroll
    for (int h = 0; h < 2; h++) {
      v16bf bfr[2];
#pragma unroll
      for (int j = 0; j < 2; j++)
        bfr[j] = frag_b(&lB[(wn * 32 + j * 16 + l15) * GSTR + h * 32 + bkb]);
#pragma unroll
      for (int i = 0; i < 4; i++) {
        const unsigned short* p = &lA[(wm * 64 + i * 16 + l15) * GSTR + h * 32 + akb];
        v16bf af = frag_a(p, p + 16);
#pragma unroll
        for (int j = 0; j < 2; j++)
          acc[i][j] = wmma_bf16(af, bfr[j], acc[i][j]);
      }
    }
  }

  // ---- epilogue ----
#pragma unroll
  for (int i = 0; i < 4; i++)
#pragma unroll
    for (int j = 0; j < 2; j++)
#pragma unroll
      for (int r = 0; r < 8; r++) {
        int mrow = m0 + wm * 64 + i * 16 + r + hl8;
        int ncol = n0 + wn * 32 + j * 16 + l15;
        float val = acc[i][j][r] + bias[ncol];
        if (EPI == 0 || EPI == 1) {
          int bidx = mrow >> 11, lpos = mrow & (Ll - 1);
          int hidx = ncol >> 7,  hd   = ncol & (HD - 1);
          if (EPI == 0) {
            if (hd < HD / 2)
              val *= (cosp[lpos * (HD / 2) + hd] - sinp[lpos * (HD / 2) + hd]);
            ((unsigned short*)Out)[(((size_t)bidx * Hh + hidx) * Ll + lpos) * HD + hd] = f2bf(val);
          } else { // V transposed: [B,H,HD,L]
            ((unsigned short*)Out)[(((size_t)bidx * Hh + hidx) * HD + hd) * Ll + lpos] = f2bf(val);
          }
        } else {
          ((float*)Out)[(size_t)mrow * Dd + ncol] = val;
        }
      }
}

// ============================================================================
// Flash attention: per (bh, 64-query tile). 4 waves x 16 query rows.
// K/V^T tiles async-staged to LDS once per 32-key step, shared by all waves.
// Q: [B,H,L,HD] bf16   K: [B,H,L,HD] bf16   VT: [B,H,HD,L] bf16
// Aout: [B,L,D] bf16 (pre-Wo attention output)
// ============================================================================
__global__ __launch_bounds__(128)
void flash_wmma(const unsigned short* __restrict__ Q,
                const unsigned short* __restrict__ K,
                const unsigned short* __restrict__ VT,
                unsigned short* __restrict__ Aout)
{
  __shared__ __align__(16) unsigned short lK[32 * KSTR];        // 32 keys x 128 hd
  __shared__ __align__(16) unsigned short lV[128 * VSTR];       // 128 hd x 32 keys
  __shared__ __align__(16) unsigned short pbuf[4][16 * VSTR];   // per-wave P tile

  const int lane = threadIdx.x & 31;
  const int wave = threadIdx.x >> 5;
  const int bh = blockIdx.y;            // 0..31
  const int q0 = blockIdx.x * 64;
  const int qrow = q0 + wave * 16;
  const int l15 = lane & 15;
  const int hl  = (lane & 16) ? 1 : 0;
  const int akb = hl ? 8 : 0;
  const int bkb = hl ? 8 : 0;

  // Q fragments (16 rows x 128 = 4 chunks of K=32), held in registers
  const unsigned short* qrp = Q + ((size_t)bh * Ll + qrow + l15) * HD;
  v16bf qf[4];
#pragma unroll
  for (int c = 0; c < 4; c++)
    qf[c] = frag_a(qrp + c * 32 + akb, qrp + c * 32 + akb + 16);

  v8f zero = {};
  v8f o[8];
#pragma unroll
  for (int j = 0; j < 8; j++) o[j] = zero;
  float m_run[8], l_run[8];
#pragma unroll
  for (int r = 0; r < 8; r++) { m_run[r] = -1e30f; l_run[r] = 0.0f; }

  const int kend = q0 + 64;             // causal: keys < q0+64 only
  for (int kb = 0; kb < kend; kb += 32) {
    // ---- async-stage K tile (32x128) and V^T tile (128x32) into LDS ----
#pragma unroll
    for (int i = 0; i < 4; i++) {       // K: 512 16B-chunks, 4 per thread
      int c = threadIdx.x + 128 * i;
      int row = c >> 4, c16 = c & 15;
      const unsigned short* g = K + ((size_t)bh * Ll + kb + row) * HD + c16 * 8;
      async_g2l_b128(lds_off(&lK[row * KSTR + c16 * 8]), g);
    }
#pragma unroll
    for (int i = 0; i < 4; i++) {       // V^T: 512 16B-chunks, 4 per thread
      int c = threadIdx.x + 128 * i;
      int row = c >> 2, c4 = c & 3;
      const unsigned short* g = VT + ((size_t)bh * HD + row) * Ll + kb + c4 * 8;
      async_g2l_b128(lds_off(&lV[row * VSTR + c4 * 8]), g);
    }
    wait_async0();
    __syncthreads();

    // ---- S = Q @ K^T for 32 keys (two 16-key column tiles) ----
    v8f st0 = zero, st1 = zero;
#pragma unroll
    for (int c = 0; c < 4; c++) {
      st0 = wmma_bf16(qf[c], frag_b(&lK[(l15)      * KSTR + c * 32 + bkb]), st0);
      st1 = wmma_bf16(qf[c], frag_b(&lK[(16 + l15) * KSTR + c * 32 + bkb]), st1);
    }

    // ---- scale, (optional) causal mask, online softmax ----
    float p0[8], p1[8], sc[8];
    auto softmax_step = [&](bool applyMask) {
#pragma unroll
      for (int r = 0; r < 8; r++) {
        float s0, s1;
        if (applyMask) {
          int qi  = qrow + r + 8 * hl;
          int k0i = kb + l15;
          s0 = (k0i      <= qi) ? st0[r] * SCALE : -1e30f;
          s1 = (k0i + 16 <= qi) ? st1[r] * SCALE : -1e30f;
        } else {
          s0 = st0[r] * SCALE;
          s1 = st1[r] * SCALE;
        }
        float t = row16_max(fmaxf(s0, s1));
        float mn  = fmaxf(m_run[r], t);
        float scl = __expf(m_run[r] - mn);
        m_run[r] = mn;
        sc[r] = scl;
        float e0 = __expf(s0 - mn);
        float e1 = __expf(s1 - mn);
        float rs = row16_sum(e0 + e1);
        l_run[r] = l_run[r] * scl + rs;
        p0[r] = e0; p1[r] = e1;
      }
    };
    if (kb + 31 <= qrow) softmax_step(false);   // wave-uniform: tile fully unmasked
    else                 softmax_step(true);    // diagonal tiles only

#pragma unroll
    for (int j = 0; j < 8; j++)
#pragma unroll
      for (int r = 0; r < 8; r++) o[j][r] *= sc[r];

    // ---- reshape P (C layout) -> A fragment via per-wave LDS buffer ----
    unsigned short* pb = pbuf[wave];
#pragma unroll
    for (int r = 0; r < 8; r++) {
      int row = r + 8 * hl;
      pb[row * VSTR + l15]      = f2bf(p0[r]);
      pb[row * VSTR + 16 + l15] = f2bf(p1[r]);
    }
    const unsigned short* pp = pb + l15 * VSTR + akb;
    v16bf pA = frag_a(pp, pp + 16);

    // ---- O += P @ V  (V^T rows are hd, keys contiguous in LDS) ----
#pragma unroll
    for (int j = 0; j < 8; j++)
      o[j] = wmma_bf16(pA, frag_b(&lV[(16 * j + l15) * VSTR + bkb]), o[j]);

    __syncthreads();   // all reads done before next step's async stores
  }

  // ---- normalize and write attention output [B,L,D] ----
  const int b = bh >> 4, h = bh & 15;
  float inv[8];
#pragma unroll
  for (int r = 0; r < 8; r++) inv[r] = 1.0f / l_run[r];
#pragma unroll
  for (int j = 0; j < 8; j++)
#pragma unroll
    for (int r = 0; r < 8; r++) {
      int row = qrow + r + 8 * hl;
      int col = h * HD + 16 * j + l15;
      Aout[((size_t)b * Ll + row) * Dd + col] = f2bf(o[j][r] * inv[r]);
    }
}

// ============================================================================
extern "C" void kernel_launch(void* const* d_in, const int* in_sizes, int n_in,
                              void* d_out, int out_size, void* d_ws, size_t ws_size,
                              hipStream_t stream) {
  (void)in_sizes; (void)n_in; (void)out_size; (void)ws_size;
  const float* x    = (const float*)d_in[0];
  const float* sinp = (const float*)d_in[1];
  const float* cosp = (const float*)d_in[2];
  // d_in[3] = causal mask (int) — applied analytically in the flash kernel
  const float* Wq = (const float*)d_in[4];  const float* bq = (const float*)d_in[5];
  const float* Wk = (const float*)d_in[6];  const float* bk = (const float*)d_in[7];
  const float* Wv = (const float*)d_in[8];  const float* bv = (const float*)d_in[9];
  const float* Wo = (const float*)d_in[10]; const float* bo = (const float*)d_in[11];
  float* out = (float*)d_out;

  const size_t qkvE = (size_t)Bb * Hh * Ll * HD;   // 8,388,608 elems
  unsigned short* qw = (unsigned short*)d_ws;
  unsigned short* kw = qw + qkvE;
  unsigned short* vw = kw + qkvE;
  unsigned short* aw = vw + qkvE;                  // [B*L, D] bf16

  dim3 gg(Dd / 128, (Bb * Ll) / 128);              // (16, 32)
  gemm_wmma<false, 0><<<gg, 256, 0, stream>>>(x,  Wq, bq, qw,  cosp, sinp);
  gemm_wmma<false, 0><<<gg, 256, 0, stream>>>(x,  Wk, bk, kw,  cosp, sinp);
  gemm_wmma<false, 1><<<gg, 256, 0, stream>>>(x,  Wv, bv, vw,  nullptr, nullptr);
  flash_wmma<<<dim3(Ll / 64, Bb * Hh), 128, 0, stream>>>(qw, kw, vw, aw);
  gemm_wmma<true, 2><<<gg, 256, 0, stream>>>(aw, Wo, bo, out, nullptr, nullptr);
}
